// ChunkedVectorizedSlidingWindowModel_80255758893218
// MI455X (gfx1250) — compile-verified, matrix operand order checked by reading, not run on hardware
//
#include <hip/hip_runtime.h>
#include <hip/hip_bf16.h>
#include <stdint.h>

typedef __attribute__((ext_vector_type(16))) _Float16 v16h;
typedef __attribute__((ext_vector_type(8)))  float    v8f;
typedef unsigned int u32x4 __attribute__((ext_vector_type(4)));
typedef int          i32x4 __attribute__((ext_vector_type(4)));
typedef int          i32x8 __attribute__((ext_vector_type(8)));

#if defined(__has_builtin)
# if __has_builtin(__builtin_amdgcn_tensor_load_to_lds) && __has_builtin(__builtin_amdgcn_s_wait_tensorcnt)
#  define HAVE_TDM 1
# endif
#endif
#ifndef HAVE_TDM
# define HAVE_TDM 0
#endif

// ---------------------------------------------------------------------------
// WMMA helpers (CDNA5 wave32, V_WMMA_F32_16X16X32_F16)
// ---------------------------------------------------------------------------
__device__ __forceinline__ v8f wmma16(v16h a, v16h b, v8f c) {
  return __builtin_amdgcn_wmma_f32_16x16x32_f16(
      /*neg_a=*/false, a, /*neg_b=*/false, b,
      /*c_mod=*/(short)0, c, /*reuse_a=*/false, /*reuse_b=*/false);
}

// Load a 16x32 f16 A/B fragment for this lane from an LDS row base pointer.
// ISA layout: dword v holds f16 pair at k = ((v&4)<<2) + (half<<3) + ((v&3)<<1)
__device__ __forceinline__ v16h load_frag(const _Float16* rowbase, int half) {
  union { unsigned u[8]; v16h h; } f;
  const unsigned short* ps = (const unsigned short*)rowbase;
#pragma unroll
  for (int v = 0; v < 8; ++v) {
    int k = ((v & 4) << 2) + (half << 3) + ((v & 3) << 1);
    f.u[v] = *(const unsigned*)(ps + k);
  }
  return f.h;
}

// ---------------------------------------------------------------------------
// Model constants
// ---------------------------------------------------------------------------
#define BATCH 2
#define SEQ   4096
#define DMODEL 1024
#define NHEAD 16
#define HDIM  64
#define WIN   128
#define NCHUNK (SEQ / WIN)   // 32
#define MROWS (BATCH * SEQ)  // 8192

// GEMM tiling
#define TM 256
#define TN 128
#define KSLICE 32
#define NKS (DMODEL / KSLICE)  // 32
#define LSTR 40   // padded LDS row stride in f16 (80B: 16B aligned, conflict-spread)

// LDS byte layout for the GEMM pipeline (two buffers: [X tile | W tile])
#define XBYTES (TM * LSTR * 2)        // 20480
#define WBYTES (TN * LSTR * 2)        // 10240
#define BUFBYTES (XBYTES + WBYTES)    // 30720 per buffer
#define GEMM_LDS (2 * BUFBYTES)       // 61440 total

// Attention LDS strides (f16 units; byte strides multiples of 16)
#define QSTR 72
#define KSTR 72
#define VSTR 264
#define PSTR 264

// ---------------------------------------------------------------------------
// Tensor Data Mover: 2D tile load (rows x cols f16) into LDS with hardware
// padding that reproduces the LSTR=40 layout (pad 4 DWORDs after every 16).
// Descriptor bitfields per CDNA5 ISA.
// Must be executed by a single wave; EXEC is ignored by TDM.
// ---------------------------------------------------------------------------
#if HAVE_TDM
__device__ __forceinline__ void tdm_load_tile(const _Float16* gptr, unsigned lds_off,
                                              unsigned tileX, unsigned tileY,
                                              unsigned tensorX, unsigned tensorY) {
  unsigned long long ga = (unsigned long long)(uintptr_t)gptr;
  u32x4 g0;
  g0[0] = 1u;                                   // count=1, user descriptor
  g0[1] = lds_off;                              // lds_addr (bytes)
  g0[2] = (unsigned)(ga & 0xFFFFFFFFu);         // global_addr[31:0]
  g0[3] = (unsigned)((ga >> 32) & 0x01FFFFFFu)  // global_addr[56:32]
        | (2u << 30);                           // type=2 ("image")
  i32x8 g1;
  g1[0] = (int)((1u << 16)     // data_size = 1 -> 2 bytes
              | (1u << 20)     // pad_enable
              | (3u << 22)     // pad_interval = 3 -> every 16 DWORDs (64B row)
              | (3u << 25));   // pad_amount  = 3 -> 4 DWORDs (8 f16)
  g1[1] = (int)((tensorX & 0xFFFFu) << 16);                       // tensor_dim0 lo
  g1[2] = (int)(((tensorX >> 16) & 0xFFFFu) | ((tensorY & 0xFFFFu) << 16));
  g1[3] = (int)(((tensorY >> 16) & 0xFFFFu) | (tileX << 16));     // tile_dim0
  g1[4] = (int)(tileY & 0xFFFFu);                                 // tile_dim1 (tile_dim2=0)
  g1[5] = (int)tensorX;                                           // tensor_dim0_stride lo
  g1[6] = 0;                                                      // stride hi / dim1_stride
  g1[7] = 0;
  i32x4 gz = {0, 0, 0, 0};
#if defined(__clang_major__) && (__clang_major__ >= 23)
  i32x8 gz8 = {0, 0, 0, 0, 0, 0, 0, 0};
  __builtin_amdgcn_tensor_load_to_lds(g0, g1, gz, gz, gz8, 0);
#else
  __builtin_amdgcn_tensor_load_to_lds(g0, g1, gz, gz, 0);
#endif
}
#endif

// ---------------------------------------------------------------------------
// f32 -> f16 conversion
// ---------------------------------------------------------------------------
__global__ void cvt_f32_f16(const float* __restrict__ in, _Float16* __restrict__ out, int n) {
  for (int i = blockIdx.x * blockDim.x + threadIdx.x; i < n; i += gridDim.x * blockDim.x)
    out[i] = (_Float16)in[i];
}

// ---------------------------------------------------------------------------
// C[M,1024] = A[M,1024] @ W^T   (W row-major [1024,1024]; B column n = W row n)
// mode 0: write f16 head-major [B,NHEAD,SEQ,HDIM] (QKV projection), scaled
// mode 1: write f32 row-major [M,1024] (output projection -> d_out)
// 256 threads = 8 waves; block tile 256x128; TDM double-buffered K pipeline.
// ---------------------------------------------------------------------------
__global__ __launch_bounds__(256) void gemm_wmma(
    const _Float16* __restrict__ A, const _Float16* __restrict__ W,
    _Float16* __restrict__ outH, float* __restrict__ outF,
    int mode, float scale) {
  // dynamic LDS starts at wave LDS offset 0; 16B alignment enables b128 DS ops
  extern __shared__ _Float16 smem[] __attribute__((aligned(16)));

  const int t = threadIdx.x;
  const int wave = t >> 5, lane = t & 31, half = lane >> 4, lr = lane & 15;
  const int mBase = blockIdx.y * TM;
  const int nBase = blockIdx.x * TN;

  v8f acc[2][8];
#pragma unroll
  for (int s = 0; s < 2; ++s)
#pragma unroll
    for (int tn = 0; tn < 8; ++tn)
      acc[s][tn] = (v8f){0.f, 0.f, 0.f, 0.f, 0.f, 0.f, 0.f, 0.f};

#if HAVE_TDM
  // Prologue: DMA slice 0 into buffer 0 (wave 0 only; TDM ignores EXEC).
  if (t == 0) {
    tdm_load_tile(A + (size_t)mBase * DMODEL, 0u, KSLICE, TM, DMODEL, MROWS);
    tdm_load_tile(W + (size_t)nBase * DMODEL, XBYTES, KSLICE, TN, DMODEL, DMODEL);
  }
  for (int ks = 0; ks < NKS; ++ks) {
    const int cur = ks & 1;
    if (t == 0) {
      if (ks + 1 < NKS) {  // DMA next slice into the other buffer
        const int k1 = (ks + 1) * KSLICE;
        const unsigned bo = (unsigned)((cur ^ 1) * BUFBYTES);
        tdm_load_tile(A + (size_t)mBase * DMODEL + k1, bo, KSLICE, TM, DMODEL, MROWS);
        tdm_load_tile(W + (size_t)nBase * DMODEL + k1, bo + XBYTES, KSLICE, TN, DMODEL, DMODEL);
        // In-order TENSORcnt: <=2 outstanding => the current slice landed.
        __builtin_amdgcn_s_wait_tensorcnt(2);
      } else {
        __builtin_amdgcn_s_wait_tensorcnt(0);
      }
    }
    __syncthreads();     // current buffer ready for every wave

    const _Float16* xs = smem + (size_t)cur * (BUFBYTES / 2);
    const _Float16* wl = xs + XBYTES / 2;
    v16h af0 = load_frag(&xs[(wave * 32 + lr) * LSTR], half);
    v16h af1 = load_frag(&xs[(wave * 32 + 16 + lr) * LSTR], half);
    v16h bf[8];
#pragma unroll
    for (int tn = 0; tn < 8; ++tn)
      bf[tn] = load_frag(&wl[(tn * 16 + lr) * LSTR], half);
#pragma unroll
    for (int tn = 0; tn < 8; ++tn) {
      acc[0][tn] = wmma16(af0, bf[tn], acc[0][tn]);
      acc[1][tn] = wmma16(af1, bf[tn], acc[1][tn]);
    }
    __syncthreads();     // all reads done before this buffer is DMA-overwritten
  }
#else
  // Fallback: synchronous cooperative staging (single buffer)
  _Float16* xs = smem;
  _Float16* wl = smem + XBYTES / 2;
  for (int ks = 0; ks < NKS; ++ks) {
    const int k0 = ks * KSLICE;
    __syncthreads();
    for (int ch = t; ch < (TM * KSLICE / 8); ch += 256) {
      int row = ch >> 2, c8 = (ch & 3) << 3;
      *(uint4*)&xs[row * LSTR + c8] =
          *(const uint4*)(A + (size_t)(mBase + row) * DMODEL + k0 + c8);
    }
    for (int ch = t; ch < (TN * KSLICE / 8); ch += 256) {
      int row = ch >> 2, c8 = (ch & 3) << 3;
      *(uint4*)&wl[row * LSTR + c8] =
          *(const uint4*)(W + (size_t)(nBase + row) * DMODEL + k0 + c8);
    }
    __syncthreads();
    v16h af0 = load_frag(&xs[(wave * 32 + lr) * LSTR], half);
    v16h af1 = load_frag(&xs[(wave * 32 + 16 + lr) * LSTR], half);
#pragma unroll
    for (int tn = 0; tn < 8; ++tn) {
      v16h bf = load_frag(&wl[(tn * 16 + lr) * LSTR], half);
      acc[0][tn] = wmma16(af0, bf, acc[0][tn]);
      acc[1][tn] = wmma16(af1, bf, acc[1][tn]);
    }
  }
#endif

#pragma unroll
  for (int sub = 0; sub < 2; ++sub)
#pragma unroll
    for (int tn = 0; tn < 8; ++tn)
#pragma unroll
      for (int r = 0; r < 8; ++r) {
        int m = mBase + wave * 32 + sub * 16 + r + 8 * half;
        int n = nBase + tn * 16 + lr;
        float val = acc[sub][tn][r] * scale;
        if (mode == 0) {
          int b = m >> 12, s = m & (SEQ - 1);
          int hh = n >> 6, d = n & (HDIM - 1);
          outH[(((size_t)(b * NHEAD + hh) * SEQ + s) << 6) + d] = (_Float16)val;
        } else {
          outF[(size_t)m * DMODEL + n] = val;
        }
      }
}

// ---------------------------------------------------------------------------
// Sliding-window attention, one block per (chunk, head, batch).
// Q already scaled by 1/sqrt(HDIM). Output written f16 to [B,S,DMODEL].
// ---------------------------------------------------------------------------
__global__ __launch_bounds__(256) void attn_swa(
    const _Float16* __restrict__ qh, const _Float16* __restrict__ kh,
    const _Float16* __restrict__ vh, _Float16* __restrict__ attn) {
  extern __shared__ _Float16 smem[] __attribute__((aligned(16)));
  _Float16* qt = smem;                 // [128][QSTR]
  _Float16* kt = qt + 128 * QSTR;      // [256][KSTR]
  _Float16* vt = kt + 256 * KSTR;      // [64][VSTR]  V transposed: [d][j]
  _Float16* pt = vt + 64 * VSTR;       // [128][PSTR] probabilities

  const int c = blockIdx.x, hh = blockIdx.y, b = blockIdx.z;
  const int t = threadIdx.x;
  const int wave = t >> 5, lane = t & 31, half = lane >> 4, lr = lane & 15;

  const size_t headBase = (size_t)(b * NHEAD + hh) * SEQ;
  const _Float16* qsrc = qh + (headBase + (size_t)c * WIN) * HDIM;
  const _Float16* ksrc = kh + (headBase + (size_t)(c - 1) * WIN) * HDIM;
  const _Float16* vsrc = vh + (headBase + (size_t)(c - 1) * WIN) * HDIM;

  // Q tile: 128x64
  for (int ch = t; ch < 1024; ch += 256) {
    int row = ch >> 3, c8 = (ch & 7) << 3;
    *(uint4*)&qt[row * QSTR + c8] = *(const uint4*)(qsrc + row * HDIM + c8);
  }
  // K tile: 256x64 (rows <128 = previous chunk; zero-filled at c==0)
  for (int ch = t; ch < 2048; ch += 256) {
    int row = ch >> 3, c8 = (ch & 7) << 3;
    uint4 val = make_uint4(0u, 0u, 0u, 0u);
    if (!(c == 0 && row < WIN)) val = *(const uint4*)(ksrc + row * HDIM + c8);
    *(uint4*)&kt[row * KSTR + c8] = val;
  }
  // V transposed into LDS: vt[d][j]
  for (int idx = t; idx < 256 * HDIM; idx += 256) {
    int j = idx >> 6, d = idx & (HDIM - 1);
    _Float16 val = (_Float16)0.f;
    if (!(c == 0 && j < WIN)) val = vsrc[j * HDIM + d];
    vt[d * VSTR + j] = val;
  }
  __syncthreads();

  // ---- Phase 1: S = Q K^T (each wave: rows [16w,16w+16), all 256 cols) ----
  v16h aq0 = load_frag(&qt[(wave * 16 + lr) * QSTR + 0], half);
  v16h aq1 = load_frag(&qt[(wave * 16 + lr) * QSTR + 32], half);
  float sreg[16][8];
#pragma unroll
  for (int tn = 0; tn < 16; ++tn) {
    v8f acc = (v8f){0.f, 0.f, 0.f, 0.f, 0.f, 0.f, 0.f, 0.f};
    v16h b0 = load_frag(&kt[(tn * 16 + lr) * KSTR + 0], half);
    v16h b1 = load_frag(&kt[(tn * 16 + lr) * KSTR + 32], half);
    acc = wmma16(aq0, b0, acc);
    acc = wmma16(aq1, b1, acc);
    int j = tn * 16 + lr;
#pragma unroll
    for (int r = 0; r < 8; ++r) {
      int i = wave * 16 + r + 8 * half;
      bool valid = (j >= i + 1) && (j <= i + WIN) && (c > 0 || j >= WIN);
      sreg[tn][r] = valid ? acc[r] : -1e30f;
    }
  }

  // ---- Softmax in registers; rows live in one 16-lane half of the wave ----
  float rinv[8];
#pragma unroll
  for (int r = 0; r < 8; ++r) {
    float m = sreg[0][r];
#pragma unroll
    for (int tn = 1; tn < 16; ++tn) m = fmaxf(m, sreg[tn][r]);
#pragma unroll
    for (int msk = 1; msk < 16; msk <<= 1) m = fmaxf(m, __shfl_xor(m, msk, 16));
    float s = 0.f;
#pragma unroll
    for (int tn = 0; tn < 16; ++tn) {
      float e = __expf(sreg[tn][r] - m);
      sreg[tn][r] = e;
      s += e;
    }
#pragma unroll
    for (int msk = 1; msk < 16; msk <<= 1) s += __shfl_xor(s, msk, 16);
    rinv[r] = 1.f / s;
  }
  // Stage P (f16) to LDS in memory row-major order for the PV matmul
#pragma unroll
  for (int tn = 0; tn < 16; ++tn) {
    int j = tn * 16 + lr;
#pragma unroll
    for (int r = 0; r < 8; ++r) {
      int i = wave * 16 + r + 8 * half;
      pt[i * PSTR + j] = (_Float16)(sreg[tn][r] * rinv[r]);
    }
  }
  __syncthreads();

  // ---- Phase 2: O = P V  (K = 256, 8 WMMA per 16x16 output tile) ----
  v16h ap[8];
#pragma unroll
  for (int ks = 0; ks < 8; ++ks)
    ap[ks] = load_frag(&pt[(wave * 16 + lr) * PSTR + ks * 32], half);
#pragma unroll
  for (int td = 0; td < 4; ++td) {
    v8f acc = (v8f){0.f, 0.f, 0.f, 0.f, 0.f, 0.f, 0.f, 0.f};
#pragma unroll
    for (int ks = 0; ks < 8; ++ks) {
      v16h bv = load_frag(&vt[(td * 16 + lr) * VSTR + ks * 32], half);
      acc = wmma16(ap[ks], bv, acc);
    }
    int d = td * 16 + lr;
#pragma unroll
    for (int r = 0; r < 8; ++r) {
      int i = wave * 16 + r + 8 * half;
      attn[((size_t)(b * SEQ + c * WIN + i)) * DMODEL + hh * HDIM + d] = (_Float16)acc[r];
    }
  }
}

// ---------------------------------------------------------------------------
// Host launch
// ---------------------------------------------------------------------------
extern "C" void kernel_launch(void* const* d_in, const int* in_sizes, int n_in,
                              void* d_out, int out_size, void* d_ws, size_t ws_size,
                              hipStream_t stream) {
  (void)in_sizes; (void)n_in; (void)out_size; (void)ws_size;
  const float* x  = (const float*)d_in[0];
  const float* Wq = (const float*)d_in[1];
  const float* Wk = (const float*)d_in[2];
  const float* Wv = (const float*)d_in[3];
  const float* Wo = (const float*)d_in[4];

  const size_t nx = (size_t)MROWS * DMODEL;   // 8388608
  const size_t nw = (size_t)DMODEL * DMODEL;  // 1048576

  _Float16* ws16 = (_Float16*)d_ws;
  _Float16* xh  = ws16;
  _Float16* wqh = xh + nx;
  _Float16* wkh = wqh + nw;
  _Float16* wvh = wkh + nw;
  _Float16* woh = wvh + nw;
  _Float16* qhp = woh + nw;   // head-major [B,H,S,64]
  _Float16* khp = qhp + nx;
  _Float16* vhp = khp + nx;
  _Float16* ah  = vhp + nx;   // [B,S,DMODEL] f16

  cvt_f32_f16<<<2048, 256, 0, stream>>>(x,  xh,  (int)nx);
  cvt_f32_f16<<<1024, 256, 0, stream>>>(Wq, wqh, (int)nw);
  cvt_f32_f16<<<1024, 256, 0, stream>>>(Wk, wkh, (int)nw);
  cvt_f32_f16<<<1024, 256, 0, stream>>>(Wv, wvh, (int)nw);
  cvt_f32_f16<<<1024, 256, 0, stream>>>(Wo, woh, (int)nw);

  dim3 ggrid(DMODEL / TN, MROWS / TM);  // (8, 32)
  gemm_wmma<<<ggrid, 256, GEMM_LDS, stream>>>(xh, wqh, qhp, nullptr, 0, 0.125f);
  gemm_wmma<<<ggrid, 256, GEMM_LDS, stream>>>(xh, wkh, khp, nullptr, 0, 1.0f);
  gemm_wmma<<<ggrid, 256, GEMM_LDS, stream>>>(xh, wvh, vhp, nullptr, 0, 1.0f);

  const size_t smem =
      (size_t)(128 * QSTR + 256 * KSTR + 64 * VSTR + 128 * PSTR) * sizeof(_Float16);
  (void)hipFuncSetAttribute(reinterpret_cast<const void*>(attn_swa),
                            hipFuncAttributeMaxDynamicSharedMemorySize, (int)smem);
  attn_swa<<<dim3(NCHUNK, NHEAD, BATCH), 256, smem, stream>>>(qhp, khp, vhp, ah);

  gemm_wmma<<<ggrid, 256, GEMM_LDS, stream>>>(ah, woh, nullptr, (float*)d_out, 1, 1.0f);
}
